// SSRupsampling_44470091383393
// MI455X (gfx1250) — compile-verified, matrix operand order checked by reading, not run on hardware
//
#include <hip/hip_runtime.h>
#include <hip/hip_bf16.h>
#include <stdint.h>

// ---------------------------------------------------------------------------
// CDNA5 (gfx1250) wave32 WMMA bf16 implementation of the SSR-upsampling chain.
// ---------------------------------------------------------------------------

typedef __attribute__((ext_vector_type(16))) __bf16 v16bf;
typedef __attribute__((ext_vector_type(8)))  float  v8f;

union FragU { uint4 q[2]; v16bf v; };

__device__ __forceinline__ unsigned short f2bf(float f) {
  union { float f; unsigned u; } c; c.f = f;
  unsigned r = c.u + 0x7FFFu + ((c.u >> 16) & 1u);
  return (unsigned short)(r >> 16);
}

__device__ __forceinline__ v8f wmma_bf16(const FragU& a, const FragU& b, v8f c) {
  return __builtin_amdgcn_wmma_f32_16x16x32_bf16(false, a.v, false, b.v,
                                                 (short)0, c, false, false);
}

// ---------------------------------------------------------------------------
// Weight prep kernels
// ---------------------------------------------------------------------------

// Wc = w_w[1024,512] @ w_channel[512,2048]  -> bf16 packed [ci/32][o:1024][ci%32]
__global__ __launch_bounds__(256) void k_fuse_wc(const float* __restrict__ w_w,
                                                 const float* __restrict__ w_ch,
                                                 unsigned short* __restrict__ WC) {
  int idx = blockIdx.x * 256 + threadIdx.x;
  if (idx >= 1024 * 2048) return;
  int o = idx >> 11, i = idx & 2047;
  float s = 0.f;
  for (int c = 0; c < 512; ++c) s += w_w[o * 512 + c] * w_ch[c * 2048 + i];
  WC[((size_t)(i >> 5) * 1024 + o) * 32 + (i & 31)] = f2bf(s);
}

// w_H[1024,512] -> bf16 packed [ci/32][o:1024][ci%32]
__global__ __launch_bounds__(256) void k_pack_wh(const float* __restrict__ wH,
                                                 unsigned short* __restrict__ WH) {
  int idx = blockIdx.x * 256 + threadIdx.x;
  if (idx >= 1024 * 512) return;
  int o = idx >> 9, i = idx & 511;
  WH[((size_t)(i >> 5) * 1024 + o) * 32 + (i & 31)] = f2bf(wH[idx]);
}

// w[co:512][ci:512][3][3] -> bf16 packed [tap:9][ci/32][co:512][ci%32]
__global__ __launch_bounds__(256) void k_pack_conv(const float* __restrict__ w,
                                                   unsigned short* __restrict__ WP) {
  int idx = blockIdx.x * 256 + threadIdx.x;
  if (idx >= 512 * 512 * 9) return;
  int t = idx;
  int kw = t % 3; t /= 3;
  int kh = t % 3; t /= 3;
  int ci = t % 512; int co = t / 512;
  int tap = kh * 3 + kw;
  WP[(((size_t)(tap * 16 + (ci >> 5)) * 512) + co) * 32 + (ci & 31)] = f2bf(w[idx]);
}

// ---------------------------------------------------------------------------
// Stage 1: fused 1x1 GEMM  Y[o=1024, p=2304] = Wc @ X   (per image n)
// Per workgroup: M-tile 256 (2 per wave), N-tile 64.  Per K-step a wave loads
// 2 A-frags (global b128) + 4 B-frags (ds b128) feeding 8 back-to-back WMMAs.
// Epilogue: pixel-shuffle along H, write bf16 XP[chunk=n*16+w/3][pix=h2*3+w%3][c2]
// ---------------------------------------------------------------------------
__global__ __launch_bounds__(256) void k_gemm_pw1(const float* __restrict__ x,
                                                  const unsigned short* __restrict__ WC,
                                                  unsigned short* __restrict__ XP) {
  __shared__ __align__(16) unsigned short Bt[64 * 32];
  int o0 = blockIdx.x * 256;   // 4 m-tiles of 256
  int p0 = blockIdx.y * 64;    // 36 pixel tiles of 2304
  int n  = blockIdx.z;         // 4 images
  int tid = threadIdx.x;
  int wave = tid >> 5, lane = tid & 31;
  int ncol = lane & 15, khalf = lane >> 4;

  v8f acc[2][4];
#pragma unroll
  for (int mi = 0; mi < 2; ++mi)
#pragma unroll
    for (int i = 0; i < 4; ++i)
#pragma unroll
      for (int j = 0; j < 8; ++j) acc[mi][i][j] = 0.f;

  const float* xb = x + (size_t)n * 2048 * 2304;
  for (int cb = 0; cb < 64; ++cb) {
    __syncthreads();
#pragma unroll
    for (int k = 0; k < 8; ++k) {            // 2048 elements, pixel-major LDS tile
      int idx = tid + k * 256;
      int ci = idx >> 6, p = idx & 63;
      Bt[p * 32 + ci] = f2bf(xb[(size_t)(cb * 32 + ci) * 2304 + p0 + p]);
    }
    __syncthreads();
    FragU a[2];
#pragma unroll
    for (int mi = 0; mi < 2; ++mi) {
      const uint4* aq = (const uint4*)(WC +
          ((size_t)(cb * 1024) + o0 + mi * 128 + wave * 16 + ncol) * 32);
      a[mi].q[0] = aq[khalf];
      a[mi].q[1] = aq[2 + khalf];
    }
    FragU b[4];
#pragma unroll
    for (int nf = 0; nf < 4; ++nf) {
      const uint4* bq = (const uint4*)(Bt + (nf * 16 + ncol) * 32);
      b[nf].q[0] = bq[khalf];
      b[nf].q[1] = bq[2 + khalf];
    }
#pragma unroll
    for (int mi = 0; mi < 2; ++mi)
#pragma unroll
      for (int nf = 0; nf < 4; ++nf)
        acc[mi][nf] = wmma_bf16(a[mi], b[nf], acc[mi][nf]);
  }

#pragma unroll
  for (int mi = 0; mi < 2; ++mi) {
    int obase = o0 + mi * 128 + wave * 16;
#pragma unroll
    for (int nf = 0; nf < 4; ++nf) {
      int p = p0 + nf * 16 + ncol;
      int h = p / 48, w_ = p % 48;
      int chunk = n * 16 + w_ / 3;
      int wl = w_ % 3;
#pragma unroll
      for (int e = 0; e < 8; ++e) {
        int o = obase + e + 8 * khalf;
        int c2 = o & 511;
        int h2 = 2 * h + (o >> 9);           // pixel-shuffle along H
        XP[((size_t)(chunk * 288 + h2 * 3 + wl)) * 512 + c2] = f2bf(acc[mi][nf][e]);
      }
    }
  }
}

// ---------------------------------------------------------------------------
// Stage 3: 1x1 GEMM with w_H over XR[n][q=h2*48+w][ci] (bf16), same tiling.
// Epilogue: pixel-shuffle along W, write bf16 XS[chunk'=n*32+h2/3][pix'=(h2%3)*96+w2][c2]
// ---------------------------------------------------------------------------
__global__ __launch_bounds__(256) void k_gemm_pw2(const unsigned short* __restrict__ XR,
                                                  const unsigned short* __restrict__ WH,
                                                  unsigned short* __restrict__ XS) {
  __shared__ __align__(16) unsigned short Bt[64 * 32];
  int o0 = blockIdx.x * 256;   // 4 m-tiles
  int p0 = blockIdx.y * 64;    // 72 pixel tiles of 4608
  int n  = blockIdx.z;
  int tid = threadIdx.x;
  int wave = tid >> 5, lane = tid & 31;
  int ncol = lane & 15, khalf = lane >> 4;

  v8f acc[2][4];
#pragma unroll
  for (int mi = 0; mi < 2; ++mi)
#pragma unroll
    for (int i = 0; i < 4; ++i)
#pragma unroll
      for (int j = 0; j < 8; ++j) acc[mi][i][j] = 0.f;

  const uint4* xrq = (const uint4*)XR;
  for (int cb = 0; cb < 16; ++cb) {
    __syncthreads();
    {                                        // 64 pixels x 64B = 256 quads
      int p = tid >> 2, part = tid & 3;
      ((uint4*)Bt)[tid] = xrq[((size_t)(n * 4608 + p0 + p)) * 64 + cb * 4 + part];
    }
    __syncthreads();
    FragU a[2];
#pragma unroll
    for (int mi = 0; mi < 2; ++mi) {
      const uint4* aq = (const uint4*)(WH +
          ((size_t)(cb * 1024) + o0 + mi * 128 + wave * 16 + ncol) * 32);
      a[mi].q[0] = aq[khalf];
      a[mi].q[1] = aq[2 + khalf];
    }
    FragU b[4];
#pragma unroll
    for (int nf = 0; nf < 4; ++nf) {
      const uint4* bq = (const uint4*)(Bt + (nf * 16 + ncol) * 32);
      b[nf].q[0] = bq[khalf];
      b[nf].q[1] = bq[2 + khalf];
    }
#pragma unroll
    for (int mi = 0; mi < 2; ++mi)
#pragma unroll
      for (int nf = 0; nf < 4; ++nf)
        acc[mi][nf] = wmma_bf16(a[mi], b[nf], acc[mi][nf]);
  }

#pragma unroll
  for (int mi = 0; mi < 2; ++mi) {
    int obase = o0 + mi * 128 + wave * 16;
#pragma unroll
    for (int nf = 0; nf < 4; ++nf) {
      int q = p0 + nf * 16 + ncol;
      int h2 = q / 48, w_ = q % 48;
#pragma unroll
      for (int e = 0; e < 8; ++e) {
        int o = obase + e + 8 * khalf;
        int c2 = o & 511;
        int w2 = 2 * w_ + (o >> 9);          // pixel-shuffle along W
        int chunk = n * 32 + h2 / 3;
        int pix = (h2 % 3) * 96 + w2;
        XS[((size_t)(chunk * 288 + pix)) * 512 + c2] = f2bf(acc[mi][nf][e]);
      }
    }
  }
}

// ---------------------------------------------------------------------------
// 3x3 conv + bias + ReLU over independent [512, R, S] chunks (pad 1), as
// implicit GEMM: 9 tap-GEMMs (M=co tile 128, N=96 pixels, K=512 in 32-blocks)
// accumulated in WMMA accumulators.  Zero-padded halo tile in LDS (pixel-major).
// Per tap: 1 A-frag + 6 batched B-frags feed 6 back-to-back WMMAs.
//   LR: S=3  (R=96), pixel tile = 32 rows x 3 cols   -> halo 34x5  = 170
//   DU: S=96 (R=3),  pixel tile = 3 rows x 32 cols   -> halo 5x34  = 170
// mode 0: write bf16 ACT layout [chunk][r*S+s][co]
// mode 1: (LR pass 2) write bf16 XR[n][q=r*48+wc*3+s][co]
// mode 2: (DU pass 2) write fp32 d_out[n][co][h=hc*3+r][w=s]
// ---------------------------------------------------------------------------
__global__ __launch_bounds__(256) void k_conv3x3(const unsigned short* __restrict__ ACT,
                                                 const unsigned short* __restrict__ WP,
                                                 const float* __restrict__ bias,
                                                 void* __restrict__ dst,
                                                 int S, int mode) {
  __shared__ __align__(16) unsigned short Bh[170 * 32];
  int chunk = blockIdx.x;
  int co0   = blockIdx.y * 128;
  int pt    = blockIdx.z;                    // 3 pixel tiles of 96
  int R  = 288 / S;
  int TS = (S == 3) ? 3 : 32;
  int TR = 96 / TS;
  int r0 = (S == 3) ? pt * 32 : 0;
  int s0 = (S == 3) ? 0 : pt * 32;
  int HS = TS + 2;                           // halo row stride
  int tid = threadIdx.x;
  int wave = tid >> 5, lane = tid & 31;
  int ncol = lane & 15, khalf = lane >> 4;
  int cobase = co0 + wave * 16;

  v8f acc[6];
#pragma unroll
  for (int i = 0; i < 6; ++i)
#pragma unroll
    for (int j = 0; j < 8; ++j) acc[i][j] = 0.f;

  const uint4* actq = (const uint4*)ACT;
  int nHalo = (TR + 2) * HS;                 // == 170

  for (int cb = 0; cb < 16; ++cb) {
    __syncthreads();
    for (int task = tid; task < nHalo * 4; task += 256) {
      int hp = task >> 2, part = task & 3;
      int rin = hp / HS, sin = hp % HS;
      int r = r0 + rin - 1, s = s0 + sin - 1;
      uint4 v = make_uint4(0u, 0u, 0u, 0u);
      if (r >= 0 && r < R && s >= 0 && s < S)
        v = actq[((size_t)(chunk * 288 + r * S + s)) * 64 + cb * 4 + part];
      ((uint4*)Bh)[hp * 4 + part] = v;
    }
    __syncthreads();
#pragma unroll
    for (int tap = 0; tap < 9; ++tap) {
      int kh = tap / 3, kw = tap % 3;
      FragU a;
      const uint4* aq =
          (const uint4*)(WP + ((size_t)((tap * 16 + cb) * 512) + cobase + ncol) * 32);
      a.q[0] = aq[khalf];
      a.q[1] = aq[2 + khalf];
      FragU b[6];
#pragma unroll
      for (int nf = 0; nf < 6; ++nf) {
        int nn = nf * 16 + ncol;
        int rl = nn / TS, sl = nn % TS;
        const uint4* bq = (const uint4*)(Bh + ((rl + kh) * HS + (sl + kw)) * 32);
        b[nf].q[0] = bq[khalf];
        b[nf].q[1] = bq[2 + khalf];
      }
#pragma unroll
      for (int nf = 0; nf < 6; ++nf)
        acc[nf] = wmma_bf16(a, b[nf], acc[nf]);
    }
  }

  float bv[8];
#pragma unroll
  for (int e = 0; e < 8; ++e) bv[e] = bias[cobase + e + 8 * khalf];

#pragma unroll
  for (int nf = 0; nf < 6; ++nf) {
    int nn = nf * 16 + ncol;
    int rl = nn / TS, sl = nn % TS;
    int r = r0 + rl, s = s0 + sl;
#pragma unroll
    for (int e = 0; e < 8; ++e) {
      int co = cobase + e + 8 * khalf;
      float v = acc[nf][e] + bv[e];
      v = v > 0.f ? v : 0.f;
      if (mode == 0) {
        ((unsigned short*)dst)[((size_t)(chunk * 288 + r * S + s)) * 512 + co] = f2bf(v);
      } else if (mode == 1) {
        int nimg = chunk >> 4, wc = chunk & 15;
        int q = r * 48 + wc * 3 + s;
        ((unsigned short*)dst)[((size_t)(nimg * 4608 + q)) * 512 + co] = f2bf(v);
      } else {
        int nimg = chunk >> 5, hc = chunk & 31;
        int h = hc * 3 + r;
        ((float*)dst)[(((size_t)nimg * 512 + co) * 96 + h) * 96 + s] = v;
      }
    }
  }
}

// ---------------------------------------------------------------------------
// Host launcher
// ---------------------------------------------------------------------------
extern "C" void kernel_launch(void* const* d_in, const int* in_sizes, int n_in,
                              void* d_out, int out_size, void* d_ws, size_t ws_size,
                              hipStream_t stream) {
  const float* x     = (const float*)d_in[0];  // [4,2048,48,48]
  const float* w_ch  = (const float*)d_in[1];  // [512,2048]
  const float* w_w   = (const float*)d_in[2];  // [1024,512]
  const float* w_H   = (const float*)d_in[3];  // [1024,512]
  const float* w_lr  = (const float*)d_in[4];  // [512,512,3,3]
  const float* b_lr  = (const float*)d_in[5];  // [512]
  const float* w_du  = (const float*)d_in[6];  // [512,512,3,3]
  const float* b_du  = (const float*)d_in[7];  // [512]
  float* out = (float*)d_out;                  // [4,512,96,96]

  char* ws = (char*)d_ws;
  size_t o_WC  = 0;                            // 2048*1024*2  = 4 MiB
  size_t o_WH  = o_WC  + (size_t)2048 * 1024 * 2;   // 512*1024*2 = 1 MiB
  size_t o_WLR = o_WH  + (size_t)512 * 1024 * 2;    // 9*512*512*2
  size_t o_WDU = o_WLR + (size_t)9 * 512 * 512 * 2;
  size_t o_A   = o_WDU + (size_t)9 * 512 * 512 * 2; // 37.75 MB ping
  size_t o_B   = o_A   + (size_t)128 * 288 * 512 * 2; // 37.75 MB pong

  unsigned short* WC  = (unsigned short*)(ws + o_WC);
  unsigned short* WH  = (unsigned short*)(ws + o_WH);
  unsigned short* WLR = (unsigned short*)(ws + o_WLR);
  unsigned short* WDU = (unsigned short*)(ws + o_WDU);
  unsigned short* BA  = (unsigned short*)(ws + o_A);  // XP / XR / XT
  unsigned short* BB  = (unsigned short*)(ws + o_B);  // XQ / XS

  // --- weight prep ---
  k_fuse_wc  <<<(1024 * 2048) / 256, 256, 0, stream>>>(w_w, w_ch, WC);
  k_pack_wh  <<<(1024 * 512) / 256, 256, 0, stream>>>(w_H, WH);
  k_pack_conv<<<(512 * 512 * 9) / 256, 256, 0, stream>>>(w_lr, WLR);
  k_pack_conv<<<(512 * 512 * 9) / 256, 256, 0, stream>>>(w_du, WDU);

  // --- stage 1: fused 1x1x2 GEMM + shuffle-H -> XP (BA) ---
  k_gemm_pw1<<<dim3(4, 36, 4), 256, 0, stream>>>(x, WC, BA);

  // --- stage 2: convLR pass1 (XP->XQ), pass2 (XQ->XR layout in BA) ---
  k_conv3x3<<<dim3(64, 4, 3), 256, 0, stream>>>(BA, WLR, b_lr, BB, 3, 0);
  k_conv3x3<<<dim3(64, 4, 3), 256, 0, stream>>>(BB, WLR, b_lr, BA, 3, 1);

  // --- stage 3: w_H GEMM + shuffle-W -> XS (BB) ---
  k_gemm_pw2<<<dim3(4, 72, 4), 256, 0, stream>>>(BA, WH, BB);

  // --- stage 4: convDU pass1 (XS->XT in BA), pass2 (XT->out fp32) ---
  k_conv3x3<<<dim3(128, 4, 3), 256, 0, stream>>>(BB, WDU, b_du, BA, 96, 0);
  k_conv3x3<<<dim3(128, 4, 3), 256, 0, stream>>>(BA, WDU, b_du, (void*)out, 96, 2);

  (void)in_sizes; (void)n_in; (void)out_size; (void)ws_size;
}
